// GNNPair_84670985273344
// MI455X (gfx1250) — matrix operand order repaired and checked by reading, not compiled, not run on hardware
//
#include <hip/hip_runtime.h>

typedef __attribute__((ext_vector_type(16))) __bf16 v16bf;
typedef __attribute__((ext_vector_type(8)))  float  v8f;

#define DH 128   // hidden dim
#define NBINS 60

// ---------------------------------------------------------------------------
// Weight repack: fp32 [K x 128] row-major -> bf16 WMMA B-tiles.
// Layout (element index t): t = ((kc*8 + nt)*32 + lane)*16 + i
//   n = nt*16 + (lane&15)
//   k = kc*32 + (lane>>4)*8 + (i<8 ? i : i+8)     (lanes 0-15: K 0..7,16..23, lanes 16-31: K 8..15,24..31)
// ---------------------------------------------------------------------------
__global__ void swizzleB_kernel(const float* __restrict__ W, __bf16* __restrict__ out,
                                int K, int totalKC) {
    int t = blockIdx.x * blockDim.x + threadIdx.x;
    int total = totalKC * 8 * 32 * 16;
    if (t >= total) return;
    int i    = t & 15;
    int lane = (t >> 4) & 31;
    int nt   = (t >> 9) & 7;
    int kc   = t >> 12;
    int k = kc * 32 + (lane >> 4) * 8 + (i < 8 ? i : i + 8);
    int n = nt * 16 + (lane & 15);
    float v = (k < K) ? W[k * DH + n] : 0.f;
    out[t] = (__bf16)v;
}

__global__ void zero_kernel(float* __restrict__ p, int total) {
    for (int i = blockIdx.x * blockDim.x + threadIdx.x; i < total;
         i += gridDim.x * blockDim.x) p[i] = 0.f;
}

__global__ void embed_kernel(const int* __restrict__ ids, const float* __restrict__ table,
                             float* __restrict__ xf, __bf16* __restrict__ xbf, int total) {
    int t = blockIdx.x * blockDim.x + threadIdx.x;
    if (t >= total) return;
    int n = t >> 7, d = t & 127;
    float v = table[ids[n] * DH + d];
    xf[t] = v; xbf[t] = (__bf16)v;
}

// ---------------------------------------------------------------------------
// m = x @ Wlin + b   (no relu).  16-row tile per wave, WMMA bf16.
// C tiles staged through a per-wave LDS slice (in-order LDS, no barrier) and
// written back as coalesced 16B stores.  nrows must be a multiple of 16.
// ---------------------------------------------------------------------------
__global__ void __launch_bounds__(256) gemm_lin_kernel(
    const __bf16* __restrict__ xbf, const __bf16* __restrict__ WB,
    const float* __restrict__ bias, __bf16* __restrict__ mbf, int nrows) {
    __shared__ __bf16 ms[8][16 * DH];
    int lane = threadIdx.x & 31, wave = threadIdx.x >> 5;
    int tile = blockIdx.x * 8 + wave;
    int r0 = tile * 16;
    if (r0 >= nrows) return;
    int m = lane & 15, hi = lane >> 4;
    int row = r0 + m;

    v16bf a[4];
#pragma unroll
    for (int kc = 0; kc < 4; kc++) {
        int kbase = kc * 32 + hi * 8;
#pragma unroll
        for (int i = 0; i < 16; i++)
            a[kc][i] = xbf[row * DH + kbase + (i < 8 ? i : i + 8)];
    }
#pragma unroll
    for (int nt = 0; nt < 8; nt++) {
        v8f c = {0.f,0.f,0.f,0.f,0.f,0.f,0.f,0.f};
#pragma unroll
        for (int kc = 0; kc < 4; kc++) {
            v16bf bm = *(const v16bf*)(WB + (((kc * 8 + nt) * 32 + lane) << 4));
            c = __builtin_amdgcn_wmma_f32_16x16x32_bf16(false, a[kc], false, bm,
                                                        (short)0, c, false, false);
        }
        int n = nt * 16 + m;
        float b = bias[n];
#pragma unroll
        for (int v = 0; v < 8; v++)
            ms[wave][(v + hi * 8) * DH + n] = (__bf16)(c[v] + b);
    }
    // coalesced write-back: 4KB tile, 128B per lane
    const uint4* src = (const uint4*)&ms[wave][0];
    uint4* dst = (uint4*)(mbf + (size_t)r0 * DH);
#pragma unroll
    for (int i = 0; i < 8; i++) dst[lane + 32 * i] = src[lane + 32 * i];
}

// ---------------------------------------------------------------------------
// Edge kernel: per 16-edge tile (one wave):
//   rbf(A,16x64 bf16) --wmma--> h1=relu(g1) --LDS--> A' --wmma--> g=relu(g2)
//   then agg[dst] += g * m[src]   (f32 atomics, L2-resident)
// E must be a multiple of 128 (8 waves x 16 edges per block).
// ---------------------------------------------------------------------------
__global__ void __launch_bounds__(256) conv_edge_kernel(
    const int* __restrict__ ei, const float* __restrict__ coords,
    const __bf16* __restrict__ g1B, const float* __restrict__ g1b,
    const __bf16* __restrict__ g2B, const float* __restrict__ g2b,
    const __bf16* __restrict__ mbf, float* __restrict__ agg, int E) {
    __shared__ __bf16 h1s[8][16 * DH];

    int lane = threadIdx.x & 31, wave = threadIdx.x >> 5;
    int tile = blockIdx.x * 8 + wave;
    int e0 = tile * 16;
    int m = lane & 15, hi = lane >> 4;
    int e = e0 + m;
    int srcReg = ei[e];
    int dstReg = ei[E + e];
    float dx = coords[dstReg * 3 + 0] - coords[srcReg * 3 + 0];
    float dy = coords[dstReg * 3 + 1] - coords[srcReg * 3 + 1];
    float dz = coords[dstReg * 3 + 2] - coords[srcReg * 3 + 2];
    float dist = sqrtf(dx * dx + dy * dy + dz * dz);

    // RBF A-matrix, 16 edges x 64 K (bins 60..63 zero pad), CDNA5 A layout
    v16bf arbf[2];
#pragma unroll
    for (int kc = 0; kc < 2; kc++) {
#pragma unroll
        for (int i = 0; i < 16; i++) {
            int k = kc * 32 + (i < 8 ? i : i + 8) + hi * 8;
            float r = 0.f;
            if (k < NBINS) { float t = dist - 0.1f * (float)k; r = __expf(-10.f * t * t); }
            arbf[kc][i] = (__bf16)r;
        }
    }

    // g1: h1 = relu(rbf @ Wg1 + b1) -> per-wave LDS slice, row-major [16][128]
#pragma unroll
    for (int nt = 0; nt < 8; nt++) {
        v8f c = {0.f,0.f,0.f,0.f,0.f,0.f,0.f,0.f};
#pragma unroll
        for (int kc = 0; kc < 2; kc++) {
            v16bf bm = *(const v16bf*)(g1B + (((kc * 8 + nt) * 32 + lane) << 4));
            c = __builtin_amdgcn_wmma_f32_16x16x32_bf16(false, arbf[kc], false, bm,
                                                        (short)0, c, false, false);
        }
        int n = nt * 16 + m;
        float bias = g1b[n];
#pragma unroll
        for (int v = 0; v < 8; v++) {
            float hv = fmaxf(c[v] + bias, 0.f);
            h1s[wave][(v + hi * 8) * DH + n] = (__bf16)hv;
        }
    }

    // reload h1 tile as A-matrix (4 K-chunks of 32); per-wave LDS is in-order
    v16bf ah[4];
#pragma unroll
    for (int kc = 0; kc < 4; kc++) {
        int kbase = kc * 32 + hi * 8;
#pragma unroll
        for (int i = 0; i < 16; i++)
            ah[kc][i] = h1s[wave][m * DH + kbase + (i < 8 ? i : i + 8)];
    }

    // g2 + gather-multiply-scatter straight from C layout
#pragma unroll
    for (int nt = 0; nt < 8; nt++) {
        v8f c = {0.f,0.f,0.f,0.f,0.f,0.f,0.f,0.f};
#pragma unroll
        for (int kc = 0; kc < 4; kc++) {
            v16bf bm = *(const v16bf*)(g2B + (((kc * 8 + nt) * 32 + lane) << 4));
            c = __builtin_amdgcn_wmma_f32_16x16x32_bf16(false, ah[kc], false, bm,
                                                        (short)0, c, false, false);
        }
        int n = nt * 16 + m;
        float bias = g2b[n];
#pragma unroll
        for (int v = 0; v < 8; v++) {
            int row = v + hi * 8;
            float g = fmaxf(c[v] + bias, 0.f);
            int s = __shfl(srcReg, row, 32);
            int t = __shfl(dstReg, row, 32);
            float mv = (float)mbf[s * DH + n];
            atomicAdd(&agg[t * DH + n], g * mv);
        }
    }
}

__global__ void residual_kernel(float* __restrict__ xf, const float* __restrict__ agg,
                                __bf16* __restrict__ xbf, int total) {
    int t = blockIdx.x * blockDim.x + threadIdx.x;
    if (t >= total) return;
    float v = fmaxf(xf[t] + agg[t], 0.f);
    xf[t] = v; xbf[t] = (__bf16)v;
}

__global__ void pool_accum_kernel(const float* __restrict__ xf, const int* __restrict__ batch,
                                  float* __restrict__ pooled, float* __restrict__ cnt, int total) {
    int t = blockIdx.x * blockDim.x + threadIdx.x;
    if (t >= total) return;
    int n = t >> 7, d = t & 127;
    int b = batch[n];
    atomicAdd(&pooled[b * DH + d], xf[t]);
    if (d == 0) atomicAdd(&cnt[b], 1.f);
}

// pooled mean -> relu(lin) -> concat emb -> L2 normalize -> hcat[:, colOff..]
__global__ void __launch_bounds__(256) branch_head_kernel(
    const float* __restrict__ pooled, const float* __restrict__ cnt,
    const float* __restrict__ Wl, const float* __restrict__ bl,
    const float* __restrict__ emb, float* __restrict__ hcat, int colOff) {
    __shared__ float p[DH];
    __shared__ float hb[1152];
    __shared__ float red[256];
    int b = blockIdx.x, tid = threadIdx.x;
    if (tid < DH) p[tid] = pooled[b * DH + tid] / fmaxf(cnt[b], 1.f);
    __syncthreads();
    if (tid < DH) {
        float acc = bl[tid];
        for (int k = 0; k < DH; k++) acc += p[k] * Wl[k * DH + tid];
        hb[tid] = fmaxf(acc, 0.f);
    }
    for (int j = tid; j < 1024; j += 256) hb[DH + j] = emb[b * 1024 + j];
    __syncthreads();
    float loc = 0.f;
    for (int i = tid; i < 1152; i += 256) loc += hb[i] * hb[i];
    red[tid] = loc; __syncthreads();
    for (int s = 128; s > 0; s >>= 1) {
        if (tid < s) red[tid] += red[tid + s];
        __syncthreads();
    }
    float nrm = fmaxf(sqrtf(red[0]), 1e-12f);
    for (int i = tid; i < 1152; i += 256) hcat[b * 2304 + colOff + i] = hb[i] / nrm;
}

// generic row-parallel dense layer; mode 0=linear 1=relu 2=sigmoid
__global__ void mlp_kernel(const float* __restrict__ in, const float* __restrict__ W,
                           const float* __restrict__ bias, float* __restrict__ out,
                           int K, int Nout, int mode) {
    extern __shared__ float row[];
    int b = blockIdx.x;
    for (int i = threadIdx.x; i < K; i += blockDim.x) row[i] = in[b * K + i];
    __syncthreads();
    for (int j = threadIdx.x; j < Nout; j += blockDim.x) {
        float acc = bias[j];
        for (int k = 0; k < K; k++) acc += row[k] * W[k * Nout + j];
        if (mode == 1) acc = fmaxf(acc, 0.f);
        else if (mode == 2) acc = 1.f / (1.f + __expf(-acc));
        out[b * Nout + j] = acc;
    }
}

// ---------------------------------------------------------------------------
extern "C" void kernel_launch(void* const* d_in, const int* in_sizes, int n_in,
                              void* d_out, int out_size, void* d_ws, size_t ws_size,
                              hipStream_t stream) {
    (void)in_sizes; (void)n_in; (void)out_size; (void)ws_size;
    const int N = 20000, E = 320000, B = 64;

    // --- input map (insertion-order flattening of setup_inputs dict) ---
    // 0: emb_res | 1+6l..6+6l: conv[l] lin.W lin.b g1.W g1.b g2.W g2.b
    // 19/20: l_lin W,b  21/22: r_lin W,b
    // 23..28: mlp W1 b1 W2 b2 W3 b3
    // 29..33: l_x, l_edge_index, l_coords, l_emb, l_batch ; 34..38: r_*
    const float* emb_res = (const float*)d_in[0];

    char* ws = (char*)d_ws;
    size_t off = 0;
    auto alloc = [&](size_t bytes) -> void* {
        void* p = ws + off;
        off = (off + bytes + 255) & ~(size_t)255;
        return p;
    };
    float*  xf     = (float*) alloc((size_t)N * DH * 4);
    __bf16* xbf    = (__bf16*)alloc((size_t)N * DH * 2);
    __bf16* mbf    = (__bf16*)alloc((size_t)N * DH * 2);
    float*  agg    = (float*) alloc((size_t)N * DH * 4);
    float*  pooled = (float*) alloc((size_t)B * DH * 4);
    float*  cnt    = (float*) alloc((size_t)B * 4);
    float*  hcat   = (float*) alloc((size_t)B * 2304 * 4);
    float*  h2     = (float*) alloc((size_t)B * 640 * 4);
    float*  h3     = (float*) alloc((size_t)B * 256 * 4);
    __bf16* linB[3]; __bf16* g1B[3]; __bf16* g2B[3];
    for (int l = 0; l < 3; l++) {
        linB[l] = (__bf16*)alloc(4 * 8 * 32 * 16 * 2);
        g1B[l]  = (__bf16*)alloc(2 * 8 * 32 * 16 * 2);
        g2B[l]  = (__bf16*)alloc(4 * 8 * 32 * 16 * 2);
    }

    // --- repack conv weights to bf16 WMMA B layout ---
    for (int l = 0; l < 3; l++) {
        swizzleB_kernel<<<(4 * 4096 + 255) / 256, 256, 0, stream>>>(
            (const float*)d_in[1 + 6 * l], linB[l], 128, 4);
        swizzleB_kernel<<<(2 * 4096 + 255) / 256, 256, 0, stream>>>(
            (const float*)d_in[3 + 6 * l], g1B[l], NBINS, 2);
        swizzleB_kernel<<<(4 * 4096 + 255) / 256, 256, 0, stream>>>(
            (const float*)d_in[5 + 6 * l], g2B[l], 128, 4);
    }

    const int totND   = N * DH;
    const int gElem   = (totND + 255) / 256;
    const int gLin    = ((N + 15) / 16 + 7) / 8;     // 16-row tiles, 8 waves/block
    const int gEdge   = ((E / 16) + 7) / 8;          // 16-edge tiles, 8 waves/block

    for (int br = 0; br < 2; br++) {
        int gi = 29 + br * 5;
        const int*   ids    = (const int*)  d_in[gi + 0];
        const int*   ei     = (const int*)  d_in[gi + 1];
        const float* coords = (const float*)d_in[gi + 2];
        const float* gemb   = (const float*)d_in[gi + 3];
        const int*   batch  = (const int*)  d_in[gi + 4];
        const float* Wl     = (const float*)d_in[19 + br * 2];
        const float* bl     = (const float*)d_in[20 + br * 2];

        embed_kernel<<<gElem, 256, 0, stream>>>(ids, emb_res, xf, xbf, totND);

        for (int l = 0; l < 3; l++) {
            const float* lb  = (const float*)d_in[2 + 6 * l];
            const float* g1b = (const float*)d_in[4 + 6 * l];
            const float* g2b = (const float*)d_in[6 + 6 * l];
            gemm_lin_kernel<<<gLin, 256, 0, stream>>>(xbf, linB[l], lb, mbf, N);
            zero_kernel<<<1024, 256, 0, stream>>>(agg, totND);
            conv_edge_kernel<<<gEdge, 256, 0, stream>>>(ei, coords, g1B[l], g1b,
                                                        g2B[l], g2b, mbf, agg, E);
            residual_kernel<<<gElem, 256, 0, stream>>>(xf, agg, xbf, totND);
        }

        zero_kernel<<<64, 256, 0, stream>>>(pooled, B * DH);
        zero_kernel<<<1, 64, 0, stream>>>(cnt, B);
        pool_accum_kernel<<<gElem, 256, 0, stream>>>(xf, batch, pooled, cnt, totND);
        branch_head_kernel<<<B, 256, 0, stream>>>(pooled, cnt, Wl, bl, gemb,
                                                  hcat, br * 1152);
    }

    mlp_kernel<<<B, 256, 2304 * 4, stream>>>(hcat, (const float*)d_in[23],
                                             (const float*)d_in[24], h2, 2304, 640, 1);
    mlp_kernel<<<B, 256, 640 * 4, stream>>>(h2, (const float*)d_in[25],
                                            (const float*)d_in[26], h3, 640, 256, 1);
    mlp_kernel<<<B, 64, 256 * 4, stream>>>(h3, (const float*)d_in[27],
                                           (const float*)d_in[28], (float*)d_out, 256, 1, 2);
}
